// DGCNN_GC_541165879370
// MI455X (gfx1250) — matrix operand order repaired and checked by reading, not compiled, not run on hardware
//
#include <hip/hip_runtime.h>

typedef __attribute__((ext_vector_type(16))) _Float16 v16h;
typedef __attribute__((ext_vector_type(8)))  _Float16 v8h;
typedef __attribute__((ext_vector_type(8)))  float    v8f;
typedef _Float16 half_t;

// ---------------- problem constants (from setup_inputs) ----------------
constexpr int B_   = 16;
constexpr int NPTS = 1024;
constexpr int KNN  = 20;
constexpr int P_   = B_ * NPTS * KNN;   // 327680 graph positions
constexpr int BN_  = B_ * NPTS;         // 16384
constexpr float EPS_ = 1e-5f;

// ---------------- workspace layout (256B aligned) ----------------
constexpr size_t O_IDX   = 0;                              // int  [B,N,20]        1,310,720 B
constexpr size_t O_WF16  = O_IDX   + (size_t)P_ * 4;       // f16 w2|w3|w4|w5        614,400 B
constexpr size_t O_W2    = O_WF16;
constexpr size_t O_W3    = O_W2 + 4096ull  * 2;
constexpr size_t O_W4    = O_W3 + 8192ull  * 2;
constexpr size_t O_W5    = O_W4 + 32768ull * 2;
constexpr size_t O_BN    = O_W5 + 262144ull * 2;           // f32 scale/shift          8,192 B
constexpr size_t O_H1    = O_BN  + 2048ull * 4;            // f16 [P,64]          41,943,040 B
constexpr size_t O_H2    = O_H1  + (size_t)P_ * 64 * 2;    // f16 [P,64]          41,943,040 B
constexpr size_t O_H3    = O_H2  + (size_t)P_ * 64 * 2;    // f16 [P,128]         83,886,080 B
constexpr size_t O_CAT32 = O_H3  + (size_t)P_ * 128 * 2;   // f32 [BN,512]        33,554,432 B
constexpr size_t O_CAT16 = O_CAT32 + (size_t)BN_ * 512 * 4;// f16 [BN,512]        16,777,216 B

// ---------------- small utility kernels ----------------
__global__ void zero_i32(int* __restrict__ p, int n) {
    for (int i = blockIdx.x * blockDim.x + threadIdx.x; i < n; i += gridDim.x * blockDim.x)
        p[i] = 0;
}

__global__ void cvt_f32_f16(const float* __restrict__ s, half_t* __restrict__ d, int n) {
    for (int i = blockIdx.x * blockDim.x + threadIdx.x; i < n; i += gridDim.x * blockDim.x)
        d[i] = (half_t)s[i];
}

__global__ void prep_bn(const float* __restrict__ g, const float* __restrict__ be,
                        const float* __restrict__ mu, const float* __restrict__ var,
                        float* __restrict__ sc, float* __restrict__ sh, int m) {
    int i = blockIdx.x * blockDim.x + threadIdx.x;
    if (i < m) {
        float s = g[i] * __frsqrt_rn(var[i] + EPS_);
        sc[i] = s;
        sh[i] = be[i] - mu[i] * s;
    }
}

// ---------------- KNN: top-20 nearest per point ----------------
__global__ __launch_bounds__(256) void knn_kernel(const float* __restrict__ x,
                                                  int* __restrict__ idx) {
    __shared__ float sx[3 * NPTS];      // 12 KB: whole cloud for this batch
    const int b = blockIdx.y;
    const int tid = threadIdx.x;
    const float* xb = x + (size_t)b * 3 * NPTS;
    for (int i = tid; i < 3 * NPTS; i += 256) sx[i] = xb[i];
    __syncthreads();

    const int n = blockIdx.x * 256 + tid;
    const float qx = sx[n], qy = sx[NPTS + n], qz = sx[2 * NPTS + n];

    float best[KNN]; int bidx[KNN];
#pragma unroll
    for (int s = 0; s < KNN; ++s) { best[s] = 3.4e38f; bidx[s] = 0; }

    for (int j = 0; j < NPTS; ++j) {
        float dx = sx[j] - qx, dy = sx[NPTS + j] - qy, dz = sx[2 * NPTS + j] - qz;
        float d = dx * dx + dy * dy + dz * dz;
        if (d < best[KNN - 1]) {            // static index -> registers
            float dd = d; int jj = j;
#pragma unroll
            for (int s = 0; s < KNN; ++s) { // stable bubble insert (ties keep lower j)
                if (dd < best[s]) {
                    float td = best[s]; int tj = bidx[s];
                    best[s] = dd; bidx[s] = jj;
                    dd = td;     jj = tj;
                }
            }
        }
    }
    const int base = (b * NPTS + n) * KNN;
#pragma unroll
    for (int s = 0; s < KNN; ++s) idx[base + s] = bidx[s];
}

// ---------------- edge feature + conv1 (64x6) + BN + ReLU + k-max ----------------
__global__ __launch_bounds__(256) void feat_conv1(const float* __restrict__ x,
                                                  const int* __restrict__ idx,
                                                  const float* __restrict__ w1,
                                                  const float* __restrict__ scale,
                                                  const float* __restrict__ shift,
                                                  half_t* __restrict__ h1,
                                                  int* __restrict__ catmax) {
    __shared__ float s_w[64 * 6];
    __shared__ float s_sc[64], s_sh[64];
    const int tid = threadIdx.x;
    if (tid < 64) { s_sc[tid] = scale[tid]; s_sh[tid] = shift[tid]; }
    for (int i = tid; i < 64 * 6; i += 256) s_w[i] = w1[i];
    __syncthreads();

    const int p = blockIdx.x * 256 + tid;          // p = ((b*N)+n)*k + kk
    const int b = p / (NPTS * KNN);
    const int rem = p % (NPTS * KNN);
    const int n = rem / KNN;
    const int j = idx[p];
    const float* xb = x + (size_t)b * 3 * NPTS;
    const float cx = xb[n], cy = xb[NPTS + n], cz = xb[2 * NPTS + n];
    const float f0 = xb[j] - cx, f1 = xb[NPTS + j] - cy, f2 = xb[2 * NPTS + j] - cz;
    const int catbase = (b * NPTS + n) * 512;      // x1 occupies channels [0,64)

    for (int og = 0; og < 64; og += 8) {
        v8h hv;
#pragma unroll
        for (int r = 0; r < 8; ++r) {
            const int o = og + r;
            const float* wr = s_w + o * 6;
            float y = f0 * wr[0] + f1 * wr[1] + f2 * wr[2]
                    + cx * wr[3] + cy * wr[4] + cz * wr[5];
            float v = y * s_sc[o] + s_sh[o];
            v = v > 0.f ? v : 0.f;
            hv[r] = (half_t)v;
            atomicMax(catmax + catbase + o, __float_as_int(v)); // ok: v >= 0
        }
        *(v8h*)(h1 + (size_t)p * 64 + og) = hv;    // h1 layout [P,64] f16
    }
}

// ---------------- WMMA fragment loads ----------------
// A (weights) [M, KD] row-major f16. A layout: lanes 0-15 carry K 0..15,
// lanes 16-31 carry K 16..31 of rows M=lane&15.
template <int KD>
__device__ __forceinline__ v16h load_a_frag(const half_t* __restrict__ wmat,
                                            int arow, int kb, int akoff) {
    const half_t* wp = wmat + (size_t)arow * KD + kb + akoff;
    v8h lo = *(const v8h*)(wp);
    v8h hi = *(const v8h*)(wp + 8);
    v16h a;
#pragma unroll
    for (int i = 0; i < 8; ++i) { a[i] = lo[i]; a[i + 8] = hi[i]; }
    return a;
}

// ---------------- software-pipelined WMMA mainloop: wave -> 32M x 64N tile ----------------
// hin [P, KD] f16 position-major: B-operand per lane is one contiguous 32B load
// (lanes 0-15 -> K 0..15, lanes 16-31 -> K 16..31 of column N=lane&15).
template <int KD>
__device__ __forceinline__ void wmma_tile32x64(const half_t* __restrict__ hin,
                                               const half_t* __restrict__ wmat,
                                               int p0, int m0, int lmod, int lhalf,
                                               v8f acc[2][4]) {
    constexpr int NIT = KD / 32;
    const int akoff = lhalf * 16;
    const half_t* hbase = hin + (size_t)(p0 + lmod) * KD + lhalf * 16;

    v16h A[2][2], Bf[2][4];                    // ping-pong fragment buffers
#pragma unroll
    for (int u = 0; u < 2; ++u) A[0][u] = load_a_frag<KD>(wmat, m0 + u * 16 + lmod, 0, akoff);
#pragma unroll
    for (int t = 0; t < 4; ++t) Bf[0][t] = *(const v16h*)(hbase + (size_t)t * 16 * KD);

#pragma unroll
    for (int it = 0; it < NIT; ++it) {
        const int cb = it & 1, nb = cb ^ 1;
        if (it + 1 < NIT) {                    // prefetch next K-step BEFORE this step's WMMAs
            const int kb = (it + 1) * 32;
#pragma unroll
            for (int u = 0; u < 2; ++u)
                A[nb][u] = load_a_frag<KD>(wmat, m0 + u * 16 + lmod, kb, akoff);
#pragma unroll
            for (int t = 0; t < 4; ++t)
                Bf[nb][t] = *(const v16h*)(hbase + (size_t)t * 16 * KD + kb);
        }
#pragma unroll
        for (int u = 0; u < 2; ++u)
#pragma unroll
            for (int t = 0; t < 4; ++t)
                acc[u][t] = __builtin_amdgcn_wmma_f32_16x16x32_f16(
                    false, A[cb][u], false, Bf[cb][t], (short)0, acc[u][t], false, false);
    }
}

// ---------------- conv2/3/4: GEMM + BN + ReLU + (h store) + fused k-max ----------------
template <int KD, bool WRITE_H>
__global__ __launch_bounds__(256) void gemm_bn_relu_max(const half_t* __restrict__ hin,
                                                        const half_t* __restrict__ wmat,
                                                        const float* __restrict__ scale,
                                                        const float* __restrict__ shift,
                                                        half_t* __restrict__ hout,
                                                        int* __restrict__ catmax,
                                                        int Mdim, int chanOff) {
    const int lane  = threadIdx.x & 31;
    const int wave  = threadIdx.x >> 5;
    const int lmod  = lane & 15;
    const int lhalf = lane >> 4;
    const int p0 = blockIdx.x * 512 + wave * 64;   // 8 waves * 64 positions
    const int m0 = blockIdx.y * 32;                // 2 x 16 output channels per wave

    v8f acc[2][4] = {};
    wmma_tile32x64<KD>(hin, wmat, p0, m0, lmod, lhalf, acc);

#pragma unroll
    for (int u = 0; u < 2; ++u) {
        const int ob = m0 + u * 16 + lhalf * 8;    // D: lane-half selects M 0..7 / 8..15
        float sc8[8], sh8[8];
#pragma unroll
        for (int r = 0; r < 8; ++r) { sc8[r] = scale[ob + r]; sh8[r] = shift[ob + r]; }
#pragma unroll
        for (int t = 0; t < 4; ++t) {
            const int p = p0 + t * 16 + lmod;
            const int n = p / KNN;                 // k innermost -> n = p / 20
            const int catbase = n * 512 + chanOff + ob;
            v8h hv;
#pragma unroll
            for (int r = 0; r < 8; ++r) {
                float v = acc[u][t][r] * sc8[r] + sh8[r];
                v = v > 0.f ? v : 0.f;
                hv[r] = (half_t)v;
                atomicMax(catmax + catbase + r, __float_as_int(v));
            }
            if constexpr (WRITE_H)
                *(v8h*)(hout + (size_t)p * Mdim + ob) = hv;
        }
    }
}

// ---------------- conv5: GEMM(512x512) + BN + ReLU -> d_out [B,512,N] fp32 ----------------
__global__ __launch_bounds__(256) void gemm5_bn_relu(const half_t* __restrict__ hin,
                                                     const half_t* __restrict__ wmat,
                                                     const float* __restrict__ scale,
                                                     const float* __restrict__ shift,
                                                     float* __restrict__ out) {
    const int lane  = threadIdx.x & 31;
    const int wave  = threadIdx.x >> 5;
    const int lmod  = lane & 15;
    const int lhalf = lane >> 4;
    const int p0 = blockIdx.x * 512 + wave * 64;   // p over B*N = 16384
    const int m0 = blockIdx.y * 32;

    v8f acc[2][4] = {};
    wmma_tile32x64<512>(hin, wmat, p0, m0, lmod, lhalf, acc);

#pragma unroll
    for (int u = 0; u < 2; ++u) {
        const int ob = m0 + u * 16 + lhalf * 8;
        float sc8[8], sh8[8];
#pragma unroll
        for (int r = 0; r < 8; ++r) { sc8[r] = scale[ob + r]; sh8[r] = shift[ob + r]; }
#pragma unroll
        for (int t = 0; t < 4; ++t) {
            const int p = p0 + t * 16 + lmod;
            const int b = p >> 10, n = p & 1023;
            float* op = out + ((size_t)b * 512 + ob) * 1024 + n;
#pragma unroll
            for (int r = 0; r < 8; ++r) {
                float v = acc[u][t][r] * sc8[r] + sh8[r];
                op[(size_t)r * 1024] = v > 0.f ? v : 0.f;
            }
        }
    }
}

// ---------------- host side ----------------
extern "C" void kernel_launch(void* const* d_in, const int* in_sizes, int n_in,
                              void* d_out, int out_size, void* d_ws, size_t ws_size,
                              hipStream_t stream) {
    (void)in_sizes; (void)n_in; (void)out_size; (void)ws_size;
    const float* x = (const float*)d_in[0];
    const float* W[5], *G[5], *BE[5], *MU[5], *VAR[5];
    for (int l = 0; l < 5; ++l) {
        W[l]   = (const float*)d_in[1 + l * 5 + 0];
        G[l]   = (const float*)d_in[1 + l * 5 + 1];
        BE[l]  = (const float*)d_in[1 + l * 5 + 2];
        MU[l]  = (const float*)d_in[1 + l * 5 + 3];
        VAR[l] = (const float*)d_in[1 + l * 5 + 4];
    }

    char* ws = (char*)d_ws;
    int*    idx   = (int*)(ws + O_IDX);
    half_t* w2h   = (half_t*)(ws + O_W2);
    half_t* w3h   = (half_t*)(ws + O_W3);
    half_t* w4h   = (half_t*)(ws + O_W4);
    half_t* w5h   = (half_t*)(ws + O_W5);
    float*  bn    = (float*)(ws + O_BN);   // per-layer [scale|shift]
    half_t* h1    = (half_t*)(ws + O_H1);
    half_t* h2    = (half_t*)(ws + O_H2);
    half_t* h3    = (half_t*)(ws + O_H3);
    float*  cat32 = (float*)(ws + O_CAT32);
    int*    cat32i= (int*)(ws + O_CAT32);
    half_t* cat16 = (half_t*)(ws + O_CAT16);

    const int   Ms[5]   = {64, 64, 128, 256, 512};
    const int   bnOff[5]= {0, 128, 256, 512, 1024};

    // 1) BN scale/shift for all layers
    for (int l = 0; l < 5; ++l)
        prep_bn<<<dim3((Ms[l] + 255) / 256), dim3(256), 0, stream>>>(
            G[l], BE[l], MU[l], VAR[l], bn + bnOff[l], bn + bnOff[l] + Ms[l], Ms[l]);

    // 2) weights -> f16 (w2..w5)
    cvt_f32_f16<<<dim3(16),   dim3(256), 0, stream>>>(W[1], w2h, 64 * 64);
    cvt_f32_f16<<<dim3(32),   dim3(256), 0, stream>>>(W[2], w3h, 128 * 64);
    cvt_f32_f16<<<dim3(128),  dim3(256), 0, stream>>>(W[3], w4h, 256 * 128);
    cvt_f32_f16<<<dim3(1024), dim3(256), 0, stream>>>(W[4], w5h, 512 * 512);

    // 3) zero the cat max-accumulator (values post-ReLU >= 0, so 0-init == identity)
    zero_i32<<<dim3(4096), dim3(256), 0, stream>>>(cat32i, BN_ * 512);

    // 4) KNN
    knn_kernel<<<dim3(NPTS / 256, B_), dim3(256), 0, stream>>>(x, idx);

    // 5) edge feature + conv1 + BN + ReLU -> h1 [P,64] f16, fused max -> cat[:,0:64]
    feat_conv1<<<dim3(P_ / 256), dim3(256), 0, stream>>>(
        x, idx, W[0], bn + bnOff[0], bn + bnOff[0] + 64, h1, cat32i);

    // 6) conv2: 64<-64, h1 -> h2, max -> cat[:,64:128]
    gemm_bn_relu_max<64, true><<<dim3(P_ / 512, 64 / 32), dim3(256), 0, stream>>>(
        h1, w2h, bn + bnOff[1], bn + bnOff[1] + 64, h2, cat32i, 64, 64);

    // 7) conv3: 128<-64, h2 -> h3, max -> cat[:,128:256]
    gemm_bn_relu_max<64, true><<<dim3(P_ / 512, 128 / 32), dim3(256), 0, stream>>>(
        h2, w3h, bn + bnOff[2], bn + bnOff[2] + 128, h3, cat32i, 128, 128);

    // 8) conv4: 256<-128, no h store (max only) -> cat[:,256:512]
    gemm_bn_relu_max<128, false><<<dim3(P_ / 512, 256 / 32), dim3(256), 0, stream>>>(
        h3, w4h, bn + bnOff[3], bn + bnOff[3] + 256, nullptr, cat32i, 256, 256);

    // 9) cat f32 -> f16 for conv5 B-operand
    cvt_f32_f16<<<dim3(8192), dim3(256), 0, stream>>>(cat32, cat16, BN_ * 512);

    // 10) conv5: 512<-512 over B*N positions -> d_out [B,512,N] fp32
    gemm5_bn_relu<<<dim3(BN_ / 512, 512 / 16 / 2), dim3(256), 0, stream>>>(
        cat16, w5h, bn + bnOff[4], bn + bnOff[4] + 512, (float*)d_out);
}
